// AttentionHead_8237747273925
// MI455X (gfx1250) — matrix operand order repaired and checked by reading, not compiled, not run on hardware
//
#include <hip/hip_runtime.h>
#include <hip/hip_bf16.h>
#include <cstdint>

typedef __attribute__((ext_vector_type(16))) _Float16 v16h;
typedef __attribute__((ext_vector_type(8)))  _Float16 v8h;
typedef __attribute__((ext_vector_type(4)))  _Float16 v4h;
typedef __attribute__((ext_vector_type(8)))  float    v8f;
typedef __attribute__((ext_vector_type(4)))  float    v4f;

#define B_ 8
#define S_ 2048
#define D_ 1024

static __device__ __forceinline__ v16h cat16(v8h lo, v8h hi) {
  return __builtin_shufflevector(lo, hi, 0,1,2,3,4,5,6,7,8,9,10,11,12,13,14,15);
}

static __device__ __forceinline__ void async_b128(unsigned lds_off, unsigned goff,
                                                  const void* sbase) {
  asm volatile("global_load_async_to_lds_b128 %0, %1, %2"
               :: "v"(lds_off), "v"(goff), "s"(sbase) : "memory");
}

// ---------------------------------------------------------------- f32 -> f16
__global__ __launch_bounds__(256) void cvt_f32_f16_kernel(
    const float* __restrict__ src, _Float16* __restrict__ dst, int n4) {
  int i = blockIdx.x * blockDim.x + threadIdx.x;
  if (i < n4) {
    v4f x = ((const v4f*)src)[i];
    v4h y;
#pragma unroll
    for (int t = 0; t < 4; ++t) y[t] = (_Float16)x[t];
    ((v4h*)dst)[i] = y;
  }
}

// ------------------------------------------------------------- WMMA GEMM
// C[m,n] = sum_k Ah[m,k] * Bm[n,k]   (f16 A row-major, f16 B rows contiguous over k)
// OMODE: 0 = f16 out [m*ldo+n] + bias
//        1 = f16 out transposed [n*ldo+m] + bias
//        2 = f32 out [m*ldo+n] = acc*scale + posb[m*ldpb+n]
//        3 = f32 out [m*ldo+n] + bias
template<int OMODE>
__global__ __launch_bounds__(128) void gemm_wmma_kernel(
    const _Float16* __restrict__ Ah, long sAz, int lda,
    const _Float16* __restrict__ Bm, long sBz, int ldb,
    const float* __restrict__ bias,
    void* __restrict__ Outv, long sOz, int ldo,
    const float* __restrict__ posb, int ldpb,
    float scale, int Kdim)
{
  constexpr int TILE_B = 128 * 32 * 2;                 // 8 KB per buffer
  // triple-buffered: one barrier per stage is sufficient
  __shared__ __attribute__((aligned(128))) _Float16 ldsB[3][128 * 32];
  __shared__ __attribute__((aligned(128))) _Float16 ldsA[3][128 * 32];

  const int z    = blockIdx.z;
  const int tid  = threadIdx.x;
  const int lane = tid & 31;
  const int wave = tid >> 5;
  const int ml   = lane & 15;
  const int h    = lane >> 4;

  const int blockN0 = blockIdx.x * 128;
  const int blockM0 = blockIdx.y * 128;
  const int n0 = blockN0 + (wave & 1) * 64;
  const int m0 = blockM0 + (wave >> 1) * 64;

  const _Float16* Az = Ah + (size_t)z * sAz;
  const _Float16* Bz = Bm + (size_t)z * sBz;

  const unsigned ldsB0 = (unsigned)(uintptr_t)(&ldsB[0][0]);
  const unsigned ldsA0 = (unsigned)(uintptr_t)(&ldsA[0][0]);

  // per-thread global row byte offsets (thread t stages row t of each tile)
  const unsigned rowB = (unsigned)(blockN0 + tid) * (unsigned)ldb * 2u;
  const unsigned rowA = (unsigned)(blockM0 + tid) * (unsigned)lda * 2u;

  v8f acc[4][4] = {};

  const int nStages = Kdim / 32;

  // issue async stage s into buffer `buf`: 8 x b128 per thread (4 B + 4 A)
  auto issue = [&](int s, int buf) {
    const unsigned kb    = (unsigned)(s * 32) * 2u;    // k byte offset
    const unsigned goffB = rowB + kb;
    const unsigned loffB = ldsB0 + (unsigned)buf * TILE_B + (unsigned)tid * 64u;
    const unsigned goffA = rowA + kb;
    const unsigned loffA = ldsA0 + (unsigned)buf * TILE_B + (unsigned)tid * 64u;
#pragma unroll
    for (int c = 0; c < 4; ++c) {
      async_b128(loffB + c * 16u, goffB + c * 16u, Bz);
      async_b128(loffA + c * 16u, goffA + c * 16u, Az);
    }
  };

  issue(0, 0);

  int cur = 0;                                         // buffer of stage s
  for (int s = 0; s < nStages; ++s) {
    const int nxt = (cur == 2) ? 0 : cur + 1;
    if (s + 1 < nStages) {
      issue(s + 1, nxt);
      asm volatile("s_wait_asynccnt 8" ::: "memory");  // stage s complete
    } else {
      asm volatile("s_wait_asynccnt 0" ::: "memory");
    }
    __syncthreads();                       // stage s resident for all waves

    // B fragments: lane = (col n=ml), k = 8*h+0..7 and 16+8*h+0..7
    v16h bf[4];
#pragma unroll
    for (int j = 0; j < 4; ++j) {
      int nl = (wave & 1) * 64 + j * 16 + ml;
      bf[j] = cat16(*(const v8h*)&ldsB[cur][nl * 32 + 8 * h], *(const v8h*)&ldsB[cur][nl * 32 + 8 * h + 16]);
    }

    // A fragments software-pipelined: load af(i+1) before WMMA block i
    const int mbase = (wave >> 1) * 64 + ml;
    const _Float16* ap0 = &ldsA[cur][mbase * 32 + 8 * h];
    v16h af = cat16(*(const v8h*)ap0, *(const v8h*)(ap0 + 16));
#pragma unroll
    for (int i = 0; i < 4; ++i) {
      v16h afn;
      if (i < 3) {
        const _Float16* ap = ap0 + (i + 1) * (16 * 32);
        afn = cat16(*(const v8h*)ap, *(const v8h*)(ap + 16));
      }
#pragma unroll
      for (int j = 0; j < 4; ++j) {
        acc[i][j] = __builtin_amdgcn_wmma_f32_16x16x32_f16(
            false, af, false, bf[j], (short)0, acc[i][j], false, false);
      }
      af = afn;
    }
    cur = nxt;
  }

  // --- epilogue: C layout = lane col n=ml, rows = h*8 + r ---
  const int rb = h * 8;
#pragma unroll
  for (int j = 0; j < 4; ++j) {
    int n = n0 + j * 16 + ml;
    float bv = 0.f;
    if (OMODE != 2 && bias != nullptr) bv = bias[n];
#pragma unroll
    for (int i = 0; i < 4; ++i) {
#pragma unroll
      for (int r = 0; r < 8; ++r) {
        int m = m0 + i * 16 + rb + r;
        float vv = acc[i][j][r];
        if constexpr (OMODE == 0) {
          (((_Float16*)Outv) + (size_t)z * sOz)[(size_t)m * ldo + n] =
              (_Float16)(vv + bv);
        } else if constexpr (OMODE == 1) {
          (((_Float16*)Outv) + (size_t)z * sOz)[(size_t)n * ldo + m] =
              (_Float16)(vv + bv);
        } else if constexpr (OMODE == 2) {
          (((float*)Outv) + (size_t)z * sOz)[(size_t)m * ldo + n] =
              vv * scale + posb[(size_t)m * ldpb + n];
        } else {
          (((float*)Outv) + (size_t)z * sOz)[(size_t)m * ldo + n] = vv + bv;
        }
      }
    }
  }
}

// ---------------------------- row softmax (writes f32 in place + f16 copy)
__global__ __launch_bounds__(256) void softmax_kernel(
    float* __restrict__ attn, _Float16* __restrict__ p16) {
  __shared__ float red[256];
  const int row = blockIdx.x;
  float*     p = attn + (size_t)row * S_;
  _Float16*  q = p16  + (size_t)row * S_;
  const int t = threadIdx.x;

  float v[8];
  float mx = -3.4e38f;
#pragma unroll
  for (int e = 0; e < 8; ++e) { v[e] = p[t + e * 256]; mx = fmaxf(mx, v[e]); }
  red[t] = mx; __syncthreads();
  for (int s = 128; s > 0; s >>= 1) {
    if (t < s) red[t] = fmaxf(red[t], red[t + s]);
    __syncthreads();
  }
  mx = red[0]; __syncthreads();

  float sum = 0.f;
#pragma unroll
  for (int e = 0; e < 8; ++e) { v[e] = __expf(v[e] - mx); sum += v[e]; }
  red[t] = sum; __syncthreads();
  for (int s = 128; s > 0; s >>= 1) {
    if (t < s) red[t] += red[t + s];
    __syncthreads();
  }
  float inv = 1.f / red[0];
#pragma unroll
  for (int e = 0; e < 8; ++e) {
    float w = v[e] * inv;
    p[t + e * 256] = w;
    q[t + e * 256] = (_Float16)w;
  }
}

// ---------------------------------------------------------------- launcher
extern "C" void kernel_launch(void* const* d_in, const int* in_sizes, int n_in,
                              void* d_out, int out_size, void* d_ws, size_t ws_size,
                              hipStream_t stream) {
  const float* qy  = (const float*)d_in[0];
  const float* ky  = (const float*)d_in[1];
  const float* vy  = (const float*)d_in[2];
  const float* Wq  = (const float*)d_in[3];
  const float* bq  = (const float*)d_in[4];
  const float* Wk  = (const float*)d_in[5];
  const float* bk  = (const float*)d_in[6];
  const float* Wv  = (const float*)d_in[7];
  const float* bv  = (const float*)d_in[8];
  const float* Wo  = (const float*)d_in[9];
  const float* bo  = (const float*)d_in[10];
  const float* pb  = (const float*)d_in[11];

  float* out  = (float*)d_out;                 // [B,S,D]
  float* attn = out + (size_t)B_ * S_ * D_;    // [B,S,S]

  const size_t NSD = (size_t)B_ * S_ * D_;     // 16.78M elements
  _Float16* Whq = (_Float16*)d_ws;             // 4 x [D,D] f16 (8 MB)
  _Float16* Whk = Whq + (size_t)D_ * D_;
  _Float16* Whv = Whk + (size_t)D_ * D_;
  _Float16* Who = Whv + (size_t)D_ * D_;
  _Float16* Qh  = Who + (size_t)D_ * D_;       // R1 [B*S, D]
  _Float16* Kh  = Qh  + NSD;                   // R2 [B*S, D]
  _Float16* Vt  = Kh  + NSD;                   // R3 [D, B*S] (transposed)
  _Float16* Oh  = Vt  + NSD;                   // R4 [B*S, D]
  _Float16* Xh  = Oh  + NSD;                   // R5 staging: converted input
  _Float16* Ph  = Qh;                          // [B*S, S] f16 probs, overlays R1+R2

  const int M  = B_ * S_;
  const dim3 bt(128);

  // 1) weights -> f16
  {
    int n4 = (D_ * D_) / 4;
    dim3 g(n4 / 256), b(256);
    cvt_f32_f16_kernel<<<g, b, 0, stream>>>(Wq, Whq, n4);
    cvt_f32_f16_kernel<<<g, b, 0, stream>>>(Wk, Whk, n4);
    cvt_f32_f16_kernel<<<g, b, 0, stream>>>(Wv, Whv, n4);
    cvt_f32_f16_kernel<<<g, b, 0, stream>>>(Wo, Who, n4);
  }

  const int nin4 = (int)(NSD / 4);
  const dim3 cg(nin4 / 256), cb(256);

  // 2) Q = query @ Wq^T + bq
  cvt_f32_f16_kernel<<<cg, cb, 0, stream>>>(qy, Xh, nin4);
  gemm_wmma_kernel<0><<<dim3(D_ / 128, M / 128, 1), bt, 0, stream>>>(
      Xh, 0, D_, Whq, 0, D_, bq, Qh, 0, D_, nullptr, 0, 1.f, D_);
  // 3) K = key @ Wk^T + bk
  cvt_f32_f16_kernel<<<cg, cb, 0, stream>>>(ky, Xh, nin4);
  gemm_wmma_kernel<0><<<dim3(D_ / 128, M / 128, 1), bt, 0, stream>>>(
      Xh, 0, D_, Whk, 0, D_, bk, Kh, 0, D_, nullptr, 0, 1.f, D_);
  // 4) V = value @ Wv^T + bv, stored transposed [D][B*S]
  cvt_f32_f16_kernel<<<cg, cb, 0, stream>>>(vy, Xh, nin4);
  gemm_wmma_kernel<1><<<dim3(D_ / 128, M / 128, 1), bt, 0, stream>>>(
      Xh, 0, D_, Whv, 0, D_, bv, Vt, 0, M, nullptr, 0, 1.f, D_);

  // 5) scores = Q @ K^T / sqrt(D) + pos_bias  (f32 to d_out attn region)
  gemm_wmma_kernel<2><<<dim3(S_ / 128, S_ / 128, B_), bt, 0, stream>>>(
      Qh, (long)S_ * D_, D_, Kh, (long)S_ * D_, D_, nullptr,
      attn, (long)S_ * S_, S_, pb, S_, 1.0f / 32.0f, D_);

  // 6) softmax rows in place + f16 copy (Ph overlays dead Qh/Kh)
  softmax_kernel<<<dim3(B_ * S_), dim3(256), 0, stream>>>(attn, Ph);

  // 7) attn_out = P @ V  (A = f16 probs, B = Vt rows over k; f16 out)
  gemm_wmma_kernel<0><<<dim3(D_ / 128, S_ / 128, B_), bt, 0, stream>>>(
      Ph, (long)S_ * S_, S_, Vt, (long)S_, M, nullptr,
      Oh, (long)S_ * D_, D_, nullptr, 0, 1.f, S_);

  // 8) output = attn_out @ Wo^T + bo  (f32 to d_out)
  gemm_wmma_kernel<3><<<dim3(D_ / 128, M / 128, 1), bt, 0, stream>>>(
      Oh, 0, D_, Who, 0, D_, bo, out, 0, D_, nullptr, 0, 1.f, D_);

  (void)in_sizes; (void)n_in; (void)out_size; (void)ws_size;
}
